// UDTransNet_65644280152598
// MI455X (gfx1250) — compile-verified, matrix operand order checked by reading, not run on hardware
//
#include <hip/hip_runtime.h>

// ---------------------------------------------------------------------------
// CDNA5 (gfx1250) UDTransNet dual-attention block.
// All matmuls -> v_wmma_f32_16x16x32_bf16 (fp32 accum), 32x32 wave tiles
// (4 WMMAs / wave / k-step), double-buffered fragment-ordered LDS so global
// (or async-DMA) traffic for tile k+1 overlaps the WMMAs of tile k.
// The bf16 P@V GEMM stages A via global_load_async_to_lds_b128 (ASYNCcnt).
// bf16 conversion uses packed native cvt; bulk bf16 passes use b128 loads.
// ---------------------------------------------------------------------------

typedef __attribute__((ext_vector_type(16))) __bf16       v16bf;
typedef __attribute__((ext_vector_type(2)))  __bf16       v2bf;
typedef __attribute__((ext_vector_type(8)))  float        v8f;
typedef __attribute__((ext_vector_type(4)))  float        v4f;
typedef __attribute__((ext_vector_type(2)))  float        v2f;
typedef __attribute__((ext_vector_type(4)))  unsigned int v4u;

__device__ __forceinline__ float bf16_bits_to_f32(unsigned short h) {
    unsigned int u = ((unsigned int)h) << 16;
    return __builtin_bit_cast(float, u);
}
__device__ __forceinline__ float loadAsF(const float* p)  { return *p; }
__device__ __forceinline__ float loadAsF(const __bf16* p) {
    return bf16_bits_to_f32(__builtin_bit_cast(unsigned short, *p));
}
__device__ __forceinline__ void storeFromF(float* p, float v)  { *p = v; }
__device__ __forceinline__ void storeFromF(__bf16* p, float v) { *p = (__bf16)v; }

template <typename T> struct IsF32        { static constexpr bool value = false; };
template <>           struct IsF32<float> { static constexpr bool value = true;  };

// packed native bf16 conversion: one v_cvt_pk_bf16_f32 per float pair
__device__ __forceinline__ unsigned pack2bf(float lo, float hi) {
    v2f f;
    f[0] = lo;
    f[1] = hi;
    const v2bf b = __builtin_convertvector(f, v2bf);
    return __builtin_bit_cast(unsigned, b);
}
__device__ __forceinline__ v4u pack8(const float* v) {
    v4u o;
#pragma unroll
    for (int j = 0; j < 4; ++j)
        o[j] = pack2bf(v[2 * j], v[2 * j + 1]);
    return o;
}

// ---------------------------------------------------------------------------
// Fragment-ordered LDS staging (ISA 7.12.2 16-bit operand layout):
//   slot[g*512 + lane*16 + i] = X[g*16 + (lane&15)][kk(i,lane>>4)]
//   kk(i,h) = (i<8 ? i+8h : i+8+8h)
// One task = 8 consecutive i of one lane = one 16B LDS b128; K-contiguous
// sources make it a contiguous 32B (fp32) / 16B (bf16) global read.
// ---------------------------------------------------------------------------
struct TaskIdx { int idx, kbase, dstoff; };
__device__ __forceinline__ TaskIdx task_decode(int t) {
    const int g      = t >> 6;
    const int w      = t & 63;
    const int lane_s = w >> 1;
    const int part   = w & 1;
    TaskIdx ti;
    ti.idx    = g * 16 + (lane_s & 15);
    ti.kbase  = part * 16 + (lane_s >> 4) * 8;
    ti.dstoff = g * 512 + lane_s * 16 + part * 8;
    return ti;
}

template <typename T, bool CONTIG, int J, int NTH>
__device__ __forceinline__ void frag_gather(const T* __restrict__ p, int ld,
                                            int idx0, int k0, int tid,
                                            float out[][8])
{
#pragma unroll
    for (int j = 0; j < J; ++j) {
        const TaskIdx ti = task_decode(tid + j * NTH);
        if (CONTIG) {
            const T* src = p + (long long)(idx0 + ti.idx) * ld + (k0 + ti.kbase);
            if (IsF32<T>::value) {
                const v4f* s4 = reinterpret_cast<const v4f*>(src);
                const v4f x0 = s4[0];
                const v4f x1 = s4[1];
#pragma unroll
                for (int e = 0; e < 4; ++e) { out[j][e] = x0[e]; out[j][4 + e] = x1[e]; }
            } else {
                const v4u raw = *reinterpret_cast<const v4u*>(src);
#pragma unroll
                for (int e = 0; e < 8; ++e)
                    out[j][e] = bf16_bits_to_f32(
                        (unsigned short)((raw[e >> 1] >> (16 * (e & 1))) & 0xffffu));
            }
        } else {
#pragma unroll
            for (int e = 0; e < 8; ++e)
                out[j][e] = loadAsF(&p[(long long)(k0 + ti.kbase + e) * ld + idx0 + ti.idx]);
        }
    }
}

template <int J, int NTH>
__device__ __forceinline__ void frag_store(__bf16* __restrict__ lds, int tid,
                                           const float in[][8])
{
#pragma unroll
    for (int j = 0; j < J; ++j) {
        const TaskIdx ti = task_decode(tid + j * NTH);
        *reinterpret_cast<v4u*>(lds + ti.dstoff) = pack8(in[j]);
    }
}

// raw 16B copies global -> fragment-ordered LDS via CDNA5 async DMA.
// Element type is a template param so non-async instantiations type-check;
// only ever executed when the source really is bf16.
template <int J, int NTH, typename T>
__device__ __forceinline__ void frag_async(const T* __restrict__ p, int ld,
                                           int idx0, int k0,
                                           __bf16* __restrict__ lds, int tid)
{
    const __bf16* pb = reinterpret_cast<const __bf16*>(p);
#pragma unroll
    for (int j = 0; j < J; ++j) {
        const TaskIdx ti = task_decode(tid + j * NTH);
        const __bf16* src = pb + (long long)(idx0 + ti.idx) * ld + (k0 + ti.kbase);
        const unsigned ldsoff =
            (unsigned)(unsigned long long)(const void*)(lds + ti.dstoff);
        const unsigned long long ga = (unsigned long long)(const void*)src;
        asm volatile("global_load_async_to_lds_b128 %0, %1, off"
                     :: "v"(ldsoff), "v"(ga) : "memory");
    }
}

// ---------------------------------------------------------------------------
// Strided-batched WMMA GEMM:  C = A @ B
//   logical A is MxK (TRA: stored A[k*lda+m]); logical B is KxN
//   (TRB: stored B[n*ldb+k]).  Batch z: base += (z/D)*s?1 + (z%D)*s?2.
// Block tile (WM*32) x (WN*32) x 32, WM*WN waves; wave tile 32x32
// (2 A-frags x 2 B-frags -> 4 WMMAs / k-step).  Double-buffered LDS.
// All launch shapes are exact multiples of the tile -> no bounds checks.
// ---------------------------------------------------------------------------
template <typename TA, typename TB, typename TC, bool TRA, bool TRB,
          int WM, int WN, bool ASYNCA>
__global__ __launch_bounds__(WM* WN * 32) void gemm_wmma(
    const TA* __restrict__ A, int lda,
    const TB* __restrict__ B, int ldb,
    TC* __restrict__ C, int ldc,
    int M, int N, int K,
    int D, long long sA1, long long sA2,
    long long sB1, long long sB2,
    long long sC1, long long sC2)
{
    constexpr int NTH = WM * WN * 32;
    constexpr int TBM = WM * 32;
    constexpr int TBN = WN * 32;
    constexpr int BK  = 32;
    constexpr int AJ  = (TBM * 4) / NTH;     // staging tasks per thread (A)
    constexpr int BJ  = (TBN * 4) / NTH;     // staging tasks per thread (B)

    const int z  = blockIdx.z;
    const int z1 = z / D, z2 = z % D;
    A += z1 * sA1 + z2 * sA2;
    B += z1 * sB1 + z2 * sB2;
    C += z1 * sC1 + z2 * sC2;

    const int m0 = blockIdx.y * TBM;
    const int n0 = blockIdx.x * TBN;

    __shared__ __bf16 Af[2][TBM * BK];
    __shared__ __bf16 Bf[2][TBN * BK];

    const int tid  = threadIdx.x;
    const int wave = tid >> 5;
    const int lane = tid & 31;
    const int half = lane >> 4;
    const int r    = lane & 15;
    const int ga0  = (wave / WN) * 2;     // first of two A fragment groups
    const int cb0  = (wave % WN) * 2;     // first of two B fragment groups

    float aReg[AJ][8];
    float bReg[BJ][8];

    v8f acc00 = {}, acc01 = {}, acc10 = {}, acc11 = {};

    // ---- prologue: stage k-tile 0 into buffer 0 ----
    if (ASYNCA) {
        frag_async<AJ, NTH>(A, lda, m0, 0, Af[0], tid);
    } else {
        frag_gather<TA, !TRA, AJ, NTH>(A, lda, m0, 0, tid, aReg);
        frag_store<AJ, NTH>(Af[0], tid, aReg);
    }
    frag_gather<TB, TRB, BJ, NTH>(B, ldb, n0, 0, tid, bReg);
    frag_store<BJ, NTH>(Bf[0], tid, bReg);
    if (ASYNCA)
        asm volatile("s_wait_asynccnt 0x0" ::: "memory");
    __syncthreads();

    int cur = 0;
    for (int k0 = 0; k0 < K; k0 += BK) {
        const bool hasNext = (k0 + BK) < K;

        // issue next tile's global traffic first (overlaps with WMMAs below)
        if (hasNext) {
            if (ASYNCA)
                frag_async<AJ, NTH>(A, lda, m0, k0 + BK, Af[cur ^ 1], tid);
            else
                frag_gather<TA, !TRA, AJ, NTH>(A, lda, m0, k0 + BK, tid, aReg);
            frag_gather<TB, TRB, BJ, NTH>(B, ldb, n0, k0 + BK, tid, bReg);
        }

        // compute on current buffer
        const v16bf* afp = reinterpret_cast<const v16bf*>(Af[cur]);
        const v16bf* bfp = reinterpret_cast<const v16bf*>(Bf[cur]);
        const v16bf a0 = afp[(ga0 + 0) * 32 + lane];
        const v16bf a1 = afp[(ga0 + 1) * 32 + lane];
        const v16bf b0 = bfp[(cb0 + 0) * 32 + lane];
        const v16bf b1 = bfp[(cb0 + 1) * 32 + lane];

        acc00 = __builtin_amdgcn_wmma_f32_16x16x32_bf16(
            false, a0, false, b0, (short)0, acc00, false, false);
        acc01 = __builtin_amdgcn_wmma_f32_16x16x32_bf16(
            false, a0, false, b1, (short)0, acc01, false, false);
        acc10 = __builtin_amdgcn_wmma_f32_16x16x32_bf16(
            false, a1, false, b0, (short)0, acc10, false, false);
        acc11 = __builtin_amdgcn_wmma_f32_16x16x32_bf16(
            false, a1, false, b1, (short)0, acc11, false, false);

        // publish next tile
        if (hasNext) {
            if (!ASYNCA)
                frag_store<AJ, NTH>(Af[cur ^ 1], tid, aReg);
            frag_store<BJ, NTH>(Bf[cur ^ 1], tid, bReg);
            if (ASYNCA)
                asm volatile("s_wait_asynccnt 0x0" ::: "memory");
        }
        __syncthreads();
        cur ^= 1;
    }

    // ---- epilogue: C/D layout VGPR i -> row i / i+8 by lane half, col lane&15
    const int colBase = n0 + (wave % WN) * 32 + r;
#pragma unroll
    for (int i = 0; i < 8; ++i) {
        const int r0 = m0 + (ga0 + 0) * 16 + i + half * 8;
        const int r1 = m0 + (ga0 + 1) * 16 + i + half * 8;
        storeFromF(&C[(long long)r0 * ldc + colBase],      acc00[i]);
        storeFromF(&C[(long long)r0 * ldc + colBase + 16], acc01[i]);
        storeFromF(&C[(long long)r1 * ldc + colBase],      acc10[i]);
        storeFromF(&C[(long long)r1 * ldc + colBase + 16], acc11[i]);
    }
}

// ---------------------------------------------------------------------------
// Per-group mean / rstd, vectorized loads (b128).  group = blockIdx.x
// ---------------------------------------------------------------------------
__global__ __launch_bounds__(256) void stats_f32(
    const float* __restrict__ in, long long count, float* __restrict__ stats,
    float eps)
{
    const long long g = blockIdx.x;
    const v4f* p = reinterpret_cast<const v4f*>(in + g * count);
    const long long n4 = count >> 2;
    float s = 0.0f, ss = 0.0f;
    for (long long i = threadIdx.x; i < n4; i += 256) {
        const v4f x = p[i];
#pragma unroll
        for (int e = 0; e < 4; ++e) { s += x[e]; ss += x[e] * x[e]; }
    }
    __shared__ float rs[256], rq[256];
    rs[threadIdx.x] = s;
    rq[threadIdx.x] = ss;
    __syncthreads();
    for (int o = 128; o > 0; o >>= 1) {
        if (threadIdx.x < o) {
            rs[threadIdx.x] += rs[threadIdx.x + o];
            rq[threadIdx.x] += rq[threadIdx.x + o];
        }
        __syncthreads();
    }
    if (threadIdx.x == 0) {
        const float mu  = rs[0] / (float)count;
        const float var = rq[0] / (float)count - mu * mu;
        stats[2 * g]     = mu;
        stats[2 * g + 1] = rsqrtf(var + eps);
    }
}

__global__ __launch_bounds__(256) void stats_bf16(
    const __bf16* __restrict__ in, long long count, float* __restrict__ stats,
    float eps)
{
    const long long g = blockIdx.x;
    const v4u* p = reinterpret_cast<const v4u*>(in + g * count);
    const long long n8 = count >> 3;
    float s = 0.0f, ss = 0.0f;
    for (long long i = threadIdx.x; i < n8; i += 256) {
        const v4u w = p[i];
#pragma unroll
        for (int e = 0; e < 8; ++e) {
            const float v = bf16_bits_to_f32(
                (unsigned short)((w[e >> 1] >> (16 * (e & 1))) & 0xffffu));
            s += v;
            ss += v * v;
        }
    }
    __shared__ float rs[256], rq[256];
    rs[threadIdx.x] = s;
    rq[threadIdx.x] = ss;
    __syncthreads();
    for (int o = 128; o > 0; o >>= 1) {
        if (threadIdx.x < o) {
            rs[threadIdx.x] += rs[threadIdx.x + o];
            rq[threadIdx.x] += rq[threadIdx.x + o];
        }
        __syncthreads();
    }
    if (threadIdx.x == 0) {
        const float mu  = rs[0] / (float)count;
        const float var = rq[0] / (float)count - mu * mu;
        stats[2 * g]     = mu;
        stats[2 * g + 1] = rsqrtf(var + eps);
    }
}

// ---------------------------------------------------------------------------
// Row softmax of inorm(x)*g+b, fp32 in/out (scalar; small rows).
// ---------------------------------------------------------------------------
__global__ __launch_bounds__(256) void softmax_f32(
    float* __restrict__ data, int rowLen, int rowsPerGroup,
    const float* __restrict__ stats,
    const float* __restrict__ gamma, const float* __restrict__ beta, int gbStride)
{
    const int row = blockIdx.x;
    const int g   = row / rowsPerGroup;
    float* p = data + (long long)row * rowLen;

    const float mu   = stats[2 * g];
    const float rstd = stats[2 * g + 1];
    const float gm   = gamma[g * gbStride];
    const float bt   = beta[g * gbStride];

    __shared__ float buf[2304];
    __shared__ float red[256];

    float lmax = -3.4e38f;
    for (int i = threadIdx.x; i < rowLen; i += 256) {
        const float y = (p[i] - mu) * rstd * gm + bt;
        buf[i] = y;
        lmax = fmaxf(lmax, y);
    }
    red[threadIdx.x] = lmax;
    __syncthreads();
    for (int o = 128; o > 0; o >>= 1) {
        if (threadIdx.x < o)
            red[threadIdx.x] = fmaxf(red[threadIdx.x], red[threadIdx.x + o]);
        __syncthreads();
    }
    const float mx = red[0];
    __syncthreads();

    float lsum = 0.0f;
    for (int i = threadIdx.x; i < rowLen; i += 256) {
        const float e = __expf(buf[i] - mx);
        buf[i] = e;
        lsum += e;
    }
    red[threadIdx.x] = lsum;
    __syncthreads();
    for (int o = 128; o > 0; o >>= 1) {
        if (threadIdx.x < o) red[threadIdx.x] += red[threadIdx.x + o];
        __syncthreads();
    }
    const float inv = 1.0f / red[0];
    for (int i = threadIdx.x; i < rowLen; i += 256)
        p[i] = buf[i] * inv;
}

// ---------------------------------------------------------------------------
// Row softmax of inorm(x)*g+b, bf16 in/out, b128 vector loads/stores.
// rowLen must be a multiple of 8.
// ---------------------------------------------------------------------------
__global__ __launch_bounds__(256) void softmax_bf16(
    __bf16* __restrict__ data, int rowLen, int rowsPerGroup,
    const float* __restrict__ stats,
    const float* __restrict__ gamma, const float* __restrict__ beta)
{
    const int row = blockIdx.x;
    const int g   = row / rowsPerGroup;
    v4u* p = reinterpret_cast<v4u*>(data + (long long)row * rowLen);
    const int n8 = rowLen >> 3;

    const float mu   = stats[2 * g];
    const float rstd = stats[2 * g + 1];
    const float gm   = gamma[g];
    const float bt   = beta[g];

    __shared__ float buf[2304];
    __shared__ float red[256];

    float lmax = -3.4e38f;
    for (int i = threadIdx.x; i < n8; i += 256) {
        const v4u w = p[i];
#pragma unroll
        for (int e = 0; e < 8; ++e) {
            const float x = bf16_bits_to_f32(
                (unsigned short)((w[e >> 1] >> (16 * (e & 1))) & 0xffffu));
            const float y = (x - mu) * rstd * gm + bt;
            buf[i * 8 + e] = y;
            lmax = fmaxf(lmax, y);
        }
    }
    red[threadIdx.x] = lmax;
    __syncthreads();
    for (int o = 128; o > 0; o >>= 1) {
        if (threadIdx.x < o)
            red[threadIdx.x] = fmaxf(red[threadIdx.x], red[threadIdx.x + o]);
        __syncthreads();
    }
    const float mx = red[0];
    __syncthreads();

    float lsum = 0.0f;
    for (int i = threadIdx.x; i < rowLen; i += 256) {
        const float e = __expf(buf[i] - mx);
        buf[i] = e;
        lsum += e;
    }
    red[threadIdx.x] = lsum;
    __syncthreads();
    for (int o = 128; o > 0; o >>= 1) {
        if (threadIdx.x < o) red[threadIdx.x] += red[threadIdx.x + o];
        __syncthreads();
    }
    const float inv = 1.0f / red[0];
    for (int i = threadIdx.x; i < n8; i += 256) {
        float v[8];
#pragma unroll
        for (int e = 0; e < 8; ++e) v[e] = buf[i * 8 + e] * inv;
        p[i] = pack8(v);
    }
}

// ---------------------------------------------------------------------------
// Host-side orchestration
// ---------------------------------------------------------------------------
static inline dim3 gemmGrid(int M, int N, int Z, int tbm, int tbn) {
    return dim3((unsigned)(N / tbn), (unsigned)(M / tbm), (unsigned)Z);
}

extern "C" void kernel_launch(void* const* d_in, const int* in_sizes, int n_in,
                              void* d_out, int out_size, void* d_ws, size_t ws_size,
                              hipStream_t stream)
{
    (void)in_sizes; (void)n_in; (void)out_size; (void)ws_size;

    const float* embC  = (const float*)d_in[4];
    const float* WqC   = (const float*)d_in[5];
    const float* WkC   = (const float*)d_in[6];
    const float* WvC   = (const float*)d_in[7];
    const float* Wq[4] = {(const float*)d_in[8],  (const float*)d_in[9],
                          (const float*)d_in[10], (const float*)d_in[11]};
    const float* Wk    = (const float*)d_in[12];
    const float* Wv    = (const float*)d_in[13];
    const float* Wo[4] = {(const float*)d_in[14], (const float*)d_in[15],
                          (const float*)d_in[16], (const float*)d_in[17]};
    const float* gamma1 = (const float*)d_in[18];
    const float* beta1  = (const float*)d_in[19];
    const float* gamma2 = (const float*)d_in[20];
    const float* beta2  = (const float*)d_in[21];
    const float* emb[4] = {(const float*)d_in[0], (const float*)d_in[1],
                           (const float*)d_in[2], (const float*)d_in[3]};

    const int B = 4, N = 576, E = 256, H = 8, C = 1024;
    const int BNrows = B * N;                 // 2304
    const int M4     = 4 * N;                 // 2304
    const long long LKV = (long long)B * M4;  // 9216
    const int dh = E / H;                     // 32

    // ---- workspace carve-out (256B aligned) ----
    char* cur = (char*)d_ws;
    auto alloc = [&](size_t bytes) {
        char* p = cur;
        cur += (bytes + 255) & ~(size_t)255;
        return p;
    };
    float*  QC   = (float*)alloc((size_t)BNrows * C * 4);
    float*  KC   = (float*)alloc((size_t)BNrows * C * 4);
    float*  VC   = (float*)alloc((size_t)BNrows * C * 4);
    float*  SIMC = (float*)alloc((size_t)B * C * C * 4);
    float*  THAT = (float*)alloc((size_t)BNrows * C * 4);
    float*  QP   = (float*)alloc((size_t)4 * BNrows * E * 4);
    float*  KP   = (float*)alloc((size_t)LKV * E * 4);
    float*  VP   = (float*)alloc((size_t)LKV * E * 4);
    float*  CTX  = (float*)alloc((size_t)4 * BNrows * E * 4);
    float*  STC  = (float*)alloc(2 * B * sizeof(float));
    float*  ST2  = (float*)alloc(2 * H * sizeof(float));
    __bf16* PB   = (__bf16*)alloc((size_t)H * N * M4 * 2);   // per-(q,b) attn chunk

    const dim3 blk128(128), blk64(64);

    // -------- stage 1: channel attention --------
    gemm_wmma<float, float, float, false, false, 2, 2, false>
        <<<gemmGrid(BNrows, C, 1, 64, 64), blk128, 0, stream>>>(
        embC, C, WqC, C, QC, C, BNrows, C, C, 1, 0, 0, 0, 0, 0, 0);
    gemm_wmma<float, float, float, false, false, 2, 2, false>
        <<<gemmGrid(BNrows, C, 1, 64, 64), blk128, 0, stream>>>(
        embC, C, WkC, C, KC, C, BNrows, C, C, 1, 0, 0, 0, 0, 0, 0);
    gemm_wmma<float, float, float, false, false, 2, 2, false>
        <<<gemmGrid(BNrows, C, 1, 64, 64), blk128, 0, stream>>>(
        embC, C, WvC, C, VC, C, BNrows, C, C, 1, 0, 0, 0, 0, 0, 0);

    // attn_c[b] = QC[b]^T @ KC[b]   (M=C, N=C, K=N)
    gemm_wmma<float, float, float, true, false, 2, 2, false>
        <<<gemmGrid(C, C, B, 64, 64), blk128, 0, stream>>>(
        QC, C, KC, C, SIMC, C, C, C, N,
        1, (long long)N * C, 0, (long long)N * C, 0, (long long)C * C, 0);

    stats_f32<<<B, 256, 0, stream>>>(SIMC, (long long)C * C, STC, 0.001f);
    softmax_f32<<<B * C, 256, 0, stream>>>(SIMC, C, C, STC, gamma1, beta1, 0);

    // T_hat[b] = VC[b] @ sim_c[b]^T   (M=N, N=C, K=C)
    gemm_wmma<float, float, float, false, true, 2, 2, false>
        <<<gemmGrid(N, C, B, 64, 64), blk128, 0, stream>>>(
        VC, C, SIMC, C, THAT, C, N, C, C,
        1, (long long)N * C, 0, (long long)C * C, 0, (long long)N * C, 0);

    // -------- stage 2: projections --------
    for (int q = 0; q < 4; ++q)
        gemm_wmma<float, float, float, false, false, 2, 2, false>
            <<<gemmGrid(BNrows, E, 1, 64, 64), blk128, 0, stream>>>(
            emb[q], E, Wq[q], E, QP + (long long)q * BNrows * E, E,
            BNrows, E, E, 1, 0, 0, 0, 0, 0, 0);

    // Kp/Vp = KV_S @ W with KV_S[b, s*N+n, e] = T_hat[b, n, s*E+e] (batch z=(b,s))
    gemm_wmma<float, float, float, false, false, 2, 2, false>
        <<<gemmGrid(N, E, B * 4, 64, 64), blk128, 0, stream>>>(
        THAT, C, Wk, E, KP, E, N, E, E,
        4, (long long)N * C, E, 0, 0, (long long)M4 * E, (long long)N * E);
    gemm_wmma<float, float, float, false, false, 2, 2, false>
        <<<gemmGrid(N, E, B * 4, 64, 64), blk128, 0, stream>>>(
        THAT, C, Wv, E, VP, E, N, E, E,
        4, (long long)N * C, E, 0, 0, (long long)M4 * E, (long long)N * E);

    // -------- stage 2: attention, chunked per (q,b); chunk lives in L2 --------
    for (int q = 0; q < 4; ++q) {
        for (int b = 0; b < B; ++b) {
            const float* Qbase = QP + (long long)q * BNrows * E + (long long)b * N * E;
            const float* Kbase = KP + (long long)b * M4 * E;
            const float* Vbase = VP + (long long)b * M4 * E;
            float*       Cbase = CTX + (long long)q * BNrows * E + (long long)b * N * E;

            // attn[h] = Qh (N x dh) @ Kh^T (dh x 4N) -> bf16  (z = h)
            gemm_wmma<float, float, __bf16, false, true, 2, 2, false>
                <<<gemmGrid(N, M4, H, 64, 64), blk128, 0, stream>>>(
                Qbase, E, Kbase, E, PB, M4, N, M4, dh,
                1, dh, 0, dh, 0, (long long)N * M4, 0);

            stats_bf16<<<H, 256, 0, stream>>>(PB, (long long)N * M4, ST2, 0.001f);
            softmax_bf16<<<H * N, 256, 0, stream>>>(PB, M4, N, ST2, gamma2, beta2);

            // ctx[h] = P (N x 4N, bf16) @ Vh (4N x dh); 64x32 tile, async A staging
            gemm_wmma<__bf16, float, float, false, false, 2, 1, true>
                <<<gemmGrid(N, dh, H, 64, 32), blk64, 0, stream>>>(
                PB, M4, Vbase, E, Cbase, E, N, dh, M4,
                1, (long long)N * M4, 0, dh, 0, dh, 0);
        }
    }

    // -------- output projections --------
    float* out = (float*)d_out;
    for (int q = 0; q < 4; ++q)
        gemm_wmma<float, float, float, false, false, 2, 2, false>
            <<<gemmGrid(BNrows, E, 1, 64, 64), blk128, 0, stream>>>(
            CTX + (long long)q * BNrows * E, E, Wo[q], E,
            out + (long long)q * BNrows * E, E,
            BNrows, E, E, 1, 0, 0, 0, 0, 0, 0);
}
